// GNN_1_interaction_46024869544459
// MI455X (gfx1250) — matrix operand order repaired and checked by reading, not compiled, not run on hardware
//
#include <hip/hip_runtime.h>

// ---------------------------------------------------------------------------
// Problem constants (match reference)
// ---------------------------------------------------------------------------
#define N_NODES   60000
#define N_EDGES   150000
#define N_GRAPHS  2048
#define DIM       512
#define NLAYERS   5
#define F_ATOM    40
#define SELF_LOOP 4
#define BN_EPS    1e-5f

typedef __attribute__((ext_vector_type(16))) _Float16     v16h;
typedef __attribute__((ext_vector_type(8)))  float        v8f;
typedef __attribute__((ext_vector_type(4)))  unsigned int u32x4;
typedef __attribute__((ext_vector_type(4)))  int          i32x4;
typedef __attribute__((ext_vector_type(8)))  int          i32x8;

// ---------------------------------------------------------------------------
// Tensor Data Mover: 1-D load of nElem f16 elements into LDS.
// D# layout per CDNA5 ISA 8.3/8.4 (group0: count/lds/global/type,
// group1: data_size, tensor/tile dims, strides).  Issued by one wave,
// completion via TENSORcnt.
// ---------------------------------------------------------------------------
__device__ inline void tdm_load_f16_1d(const _Float16* src, void* ldsDst,
                                       int nElem) {
    uint64_t ga = (uint64_t)(uintptr_t)src;
    uint32_t lo = (uint32_t)(uintptr_t)ldsDst;   // LDS byte offset (addr[31:0])

    u32x4 g0;
    g0[0] = 1u;                                   // count=1, is_restore=0
    g0[1] = lo;                                   // lds_addr     (bits 63:32)
    g0[2] = (uint32_t)ga;                         // global_addr  (bits 95:64)
    g0[3] = (uint32_t)((ga >> 32) & 0x1FFFFFFu)   // global_addr[56:32]
            | (2u << 30);                         // type = 2 ("image")

    uint32_t td0 = (uint32_t)nElem;               // tensor_dim0
    uint32_t td1 = 1u;                            // tensor_dim1
    uint32_t tl0 = (uint32_t)nElem;               // tile_dim0 (<= 8192, fits 16b)
    uint32_t tl1 = 1u;                            // tile_dim1
    uint64_t st0 = (uint64_t)nElem;               // tensor_dim0_stride

    i32x8 g1;
    g1[0] = 0x00010000;                           // wg_mask=0, data_size=1 (2B)
    g1[1] = (int)((td0 & 0xFFFFu) << 16);         // tensor_dim0[15:0] @ 63:48
    g1[2] = (int)(((td0 >> 16) & 0xFFFFu) | ((td1 & 0xFFFFu) << 16));
    g1[3] = (int)(((td1 >> 16) & 0xFFFFu) | ((tl0 & 0xFFFFu) << 16));
    g1[4] = (int)(tl1 & 0xFFFFu);                 // tile_dim1 | tile_dim2=0
    g1[5] = (int)(uint32_t)(st0 & 0xFFFFFFFFu);   // stride0[31:0]
    g1[6] = (int)(uint32_t)((st0 >> 32) & 0xFFFFu); // stride0[47:32] | stride1=0
    g1[7] = 0;

    i32x4 z4 = {0, 0, 0, 0};
#if __clang_major__ >= 23
    i32x8 z8 = {0, 0, 0, 0, 0, 0, 0, 0};
    __builtin_amdgcn_tensor_load_to_lds(g0, g1, z4, z4, z8, 0);
#else
    __builtin_amdgcn_tensor_load_to_lds(g0, g1, z4, z4, 0);
#endif
}

// ---------------------------------------------------------------------------
// Pack f32 weight matrix (K x N row-major) into WMMA B-fragment order, f16.
// Chunk (kb, ns) holds 32 lanes x 16 f16 contiguous:
//   P[((kb*(N/16) + ns)*512) + lane*16 + i] = B[(kb*32 + (lane&16) + i)*N
//                                               + ns*16 + (lane&15)]
// so each lane's 16-element fragment is one contiguous 32-byte read.
// ---------------------------------------------------------------------------
__global__ void pack_w_f16_kernel(const float* __restrict__ B,
                                  _Float16* __restrict__ P, int K, int N) {
    int total = K * N;
    int nsCount = N >> 4;
    for (int idx = blockIdx.x * blockDim.x + threadIdx.x; idx < total;
         idx += gridDim.x * blockDim.x) {
        int chunk  = idx >> 9;
        int within = idx & 511;
        int lane = within >> 4;
        int i    = within & 15;
        int kb = chunk / nsCount;
        int ns = chunk - kb * nsCount;
        int k = (kb << 5) + (lane & 16) + i;
        int n = (ns << 4) + (lane & 15);
        P[idx] = (_Float16)B[(size_t)k * N + n];
    }
}

// ---------------------------------------------------------------------------
// zero-fill (grid-stride)
// ---------------------------------------------------------------------------
__global__ void zero_kernel(float* __restrict__ p, int n) {
    for (int i = blockIdx.x * blockDim.x + threadIdx.x; i < n;
         i += gridDim.x * blockDim.x)
        p[i] = 0.0f;
}

// ---------------------------------------------------------------------------
// Small-K dense layer (K=40): Y = relu(X @ W + b).  One thread per output.
// ---------------------------------------------------------------------------
__global__ void dense_smallk_relu_kernel(const float* __restrict__ X,
                                         const float* __restrict__ W,
                                         const float* __restrict__ b,
                                         float* __restrict__ Y,
                                         int M, int N, int K) {
    int idx = blockIdx.x * blockDim.x + threadIdx.x;
    if (idx >= M * N) return;
    int row = idx / N, col = idx - row * N;
    float s = b[col];
    const float* xr = X + (size_t)row * K;
    for (int k = 0; k < K; ++k) s = fmaf(xr[k], W[(size_t)k * N + col], s);
    Y[idx] = fmaxf(s, 0.0f);
}

// ---------------------------------------------------------------------------
// agg = h + edge_emb1[l][SELF_LOOP] + edge_emb2[l][0]
// ---------------------------------------------------------------------------
__global__ void agg_init_kernel(const float* __restrict__ h,
                                const float* __restrict__ e1l,
                                const float* __restrict__ e2l,
                                float* __restrict__ agg, int total) {
    for (int i = blockIdx.x * blockDim.x + threadIdx.x; i < total;
         i += gridDim.x * blockDim.x) {
        int d = i & (DIM - 1);
        agg[i] = h[i] + e1l[SELF_LOOP * DIM + d] + e2l[d];
    }
}

// ---------------------------------------------------------------------------
// Edge scatter:  agg[dst] += h[src] + e1[bond_type] + e2[bond_dir]
// ---------------------------------------------------------------------------
__global__ __launch_bounds__(256)
void edge_scatter_kernel(const float* __restrict__ h,
                         const int* __restrict__ src,
                         const int* __restrict__ dst,
                         const int* __restrict__ attr,
                         const float* __restrict__ e1l,
                         const float* __restrict__ e2l,
                         float* __restrict__ agg) {
    int e  = blockIdx.x;
    int s  = src[e];
    int d  = dst[e];
    int bt = attr[2 * e + 0];
    int bd = attr[2 * e + 1];
    const float* hs = h + (size_t)s * DIM;
    const float* t1 = e1l + (size_t)bt * DIM;
    const float* t2 = e2l + (size_t)bd * DIM;
    float* out = agg + (size_t)d * DIM;
    for (int c = threadIdx.x; c < DIM; c += blockDim.x)
        unsafeAtomicAdd(&out[c], hs[c] + t1[c] + t2[c]);
}

// ---------------------------------------------------------------------------
// WMMA GEMM with TDM-fed, double-buffered LDS weight tiles.
//   C(M x N) = [relu]( A(M x K, f32, lda) @ B(K x N, pre-packed f16) + bias )
// Block = 128 threads (4 waves).  Wave w covers rows [tileM,tileM+16) and
// TN 16-col subtiles; block covers 64*TN columns.  Wave 0 drives the TDM.
// A fragment converted in-register and reused across TN WMMAs.
// ---------------------------------------------------------------------------
template <int TN>
__global__ __launch_bounds__(128)
void gemm_wmma_tdm_kernel(const float* __restrict__ A, int lda,
                          const _Float16* __restrict__ Bp,
                          const float* __restrict__ bias,
                          float* __restrict__ C, int ldc,
                          int N, int K, int doRelu) {
    constexpr int SUBS  = 4 * TN;   // 16-col subtiles per block
    constexpr int CHUNK = 512;      // f16 per (kb, subtile) fragment chunk
    __shared__ _Float16 shB[2][SUBS * CHUNK];

    const int lane  = threadIdx.x & 31;
    const int wave  = threadIdx.x >> 5;
    const int tileM = blockIdx.x * 16;
    const int s0    = blockIdx.y * SUBS;   // first subtile handled by block
    const int nsCnt = N >> 4;
    const int KB    = K >> 5;

    const int mn     = lane & 15;          // A row / B-C column within tile
    const int aKB    = (lane & 16);        // A K-group base: 0 or 16
    const int rowOff = (lane & 16) >> 1;   // C row offset:   0 or 8
    const float* Arow = A + (size_t)(tileM + mn) * lda;

    v8f acc[TN];
#pragma unroll
    for (int t = 0; t < TN; ++t) acc[t] = (v8f){};

    if (wave == 0)
        tdm_load_f16_1d(Bp + (size_t)s0 * CHUNK, (void*)&shB[0][0], SUBS * CHUNK);

    for (int kb = 0; kb < KB; ++kb) {
        if (wave == 0) __builtin_amdgcn_s_wait_tensorcnt(0);
        __syncthreads();                       // buffer kb&1 ready for all waves
        if (wave == 0 && kb + 1 < KB)
            tdm_load_f16_1d(Bp + ((size_t)(kb + 1) * nsCnt + s0) * CHUNK,
                            (void*)&shB[(kb + 1) & 1][0], SUBS * CHUNK);

        // A fragment (contiguous, matches packed B): lanes 0-15 K {0..15}, 16-31 {16..31}
        const int k0 = kb << 5;
        v16h aF;
#pragma unroll
        for (int i = 0; i < 8; ++i) {
            aF[i]     = (_Float16)Arow[k0 + aKB + i];
            aF[i + 8] = (_Float16)Arow[k0 + aKB + 8 + i];
        }

        const _Float16* bBase = &shB[kb & 1][(wave * TN) * CHUNK + lane * 16];
#pragma unroll
        for (int t = 0; t < TN; ++t) {
            v16h bF = *(const v16h*)(bBase + t * CHUNK);   // one 32B LDS read
            acc[t] = __builtin_amdgcn_wmma_f32_16x16x32_f16(
                false, aF, false, bF, (short)0, acc[t], false, false);
        }
        __syncthreads();                       // done reading before overwrite
    }

#pragma unroll
    for (int t = 0; t < TN; ++t) {
        int col = (s0 + wave * TN + t) * 16 + mn;
        float bv = bias[col];
#pragma unroll
        for (int v = 0; v < 8; ++v) {
            float val = acc[t][v] + bv;
            if (doRelu) val = fmaxf(val, 0.0f);
            C[(size_t)(tileM + rowOff + v) * ldc + col] = val;
        }
    }
}

// ---------------------------------------------------------------------------
// BatchNorm statistics: one block per column, reduce over M rows.
// ---------------------------------------------------------------------------
__global__ __launch_bounds__(256)
void bn_stats_kernel(const float* __restrict__ h2, float* __restrict__ stats,
                     int M, int N) {
    const int d = blockIdx.x;
    float s = 0.0f, s2 = 0.0f;
    for (int r = threadIdx.x; r < M; r += blockDim.x) {
        float v = h2[(size_t)r * N + d];
        s += v; s2 += v * v;
    }
    __shared__ float shs[256], shq[256];
    shs[threadIdx.x] = s; shq[threadIdx.x] = s2;
    __syncthreads();
    for (int st = 128; st > 0; st >>= 1) {
        if (threadIdx.x < st) {
            shs[threadIdx.x] += shs[threadIdx.x + st];
            shq[threadIdx.x] += shq[threadIdx.x + st];
        }
        __syncthreads();
    }
    if (threadIdx.x == 0) {
        float mu  = shs[0] / (float)M;
        float var = shq[0] / (float)M - mu * mu;
        stats[d]     = mu;
        stats[N + d] = var;
    }
}

__global__ void bn_apply_kernel(const float* __restrict__ h2,
                                const float* __restrict__ stats,
                                const float* __restrict__ gamma,
                                const float* __restrict__ beta,
                                float* __restrict__ h, int total, int doRelu) {
    for (int i = blockIdx.x * blockDim.x + threadIdx.x; i < total;
         i += gridDim.x * blockDim.x) {
        int d = i & (DIM - 1);
        float mu  = stats[d];
        float var = stats[DIM + d];
        float v = gamma[d] * (h2[i] - mu) * rsqrtf(var + BN_EPS) + beta[d];
        h[i] = doRelu ? fmaxf(v, 0.0f) : v;
    }
}

// ---------------------------------------------------------------------------
// Segment pooling into concat buffer columns [0,512)
// ---------------------------------------------------------------------------
__global__ __launch_bounds__(256)
void pool_kernel(const float* __restrict__ h, const int* __restrict__ batch,
                 float* __restrict__ zcat) {
    int n = blockIdx.x;
    int g = batch[n];
    const float* hr = h + (size_t)n * DIM;
    float* out = zcat + (size_t)g * (2 * DIM);
    for (int c = threadIdx.x; c < DIM; c += blockDim.x)
        unsafeAtomicAdd(&out[c], hr[c]);
}

// ---------------------------------------------------------------------------
// Final projection: out[g] = z[g] @ last_W + last_b   (K=128, N=1)
// ---------------------------------------------------------------------------
__global__ void final_dot_kernel(const float* __restrict__ z,
                                 const float* __restrict__ w,
                                 const float* __restrict__ b,
                                 float* __restrict__ out, int M, int K) {
    int g = blockIdx.x * blockDim.x + threadIdx.x;
    if (g >= M) return;
    float s = b[0];
    const float* zr = z + (size_t)g * K;
    for (int k = 0; k < K; ++k) s = fmaf(zr[k], w[k], s);
    out[g] = s;
}

// ---------------------------------------------------------------------------
// Host orchestration
// ---------------------------------------------------------------------------
extern "C" void kernel_launch(void* const* d_in, const int* in_sizes, int n_in,
                              void* d_out, int out_size, void* d_ws, size_t ws_size,
                              hipStream_t stream) {
    (void)in_sizes; (void)n_in; (void)out_size; (void)ws_size;

    const float* solute_x   = (const float*)d_in[0];
    const float* solvent_x  = (const float*)d_in[1];
    const int*   edge_index = (const int*)d_in[2];
    const int*   edge_attr  = (const int*)d_in[3];
    const int*   batch      = (const int*)d_in[4];
    const float* x_emb_W    = (const float*)d_in[5];
    const float* x_emb_b    = (const float*)d_in[6];
    const float* mlp1_W     = (const float*)d_in[7];
    const float* mlp1_b     = (const float*)d_in[8];
    const float* mlp2_W     = (const float*)d_in[9];
    const float* mlp2_b     = (const float*)d_in[10];
    const float* edge_emb1  = (const float*)d_in[11];
    const float* edge_emb2  = (const float*)d_in[12];
    const float* bn_gamma   = (const float*)d_in[13];
    const float* bn_beta    = (const float*)d_in[14];
    const float* sW1        = (const float*)d_in[15];
    const float* sb1        = (const float*)d_in[16];
    const float* sW2        = (const float*)d_in[17];
    const float* sb2        = (const float*)d_in[18];
    const float* out_W0     = (const float*)d_in[19];
    const float* out_b0     = (const float*)d_in[20];
    const float* out_W1     = (const float*)d_in[21];
    const float* out_b1     = (const float*)d_in[22];
    const float* out_W2     = (const float*)d_in[23];
    const float* out_b2     = (const float*)d_in[24];
    const float* last_W     = (const float*)d_in[25];
    const float* last_b     = (const float*)d_in[26];

    const int* e_src = edge_index;
    const int* e_dst = edge_index + N_EDGES;

    // ---- workspace carve -------------------------------------------------
    char*  base = (char*)d_ws;
    size_t off  = 0;
    auto carve = [&](size_t bytes) -> void* {
        void* p = base + off;
        off += (bytes + 255) & ~(size_t)255;
        return p;
    };
    float*    h    = (float*)carve((size_t)N_NODES * DIM * 4);
    float*    agg  = (float*)carve((size_t)N_NODES * DIM * 4);      // reused as h2
    float*    mid  = (float*)carve((size_t)N_NODES * 2 * DIM * 4);
    float*    stats= (float*)carve(2 * DIM * 4);
    float*    zcat = (float*)carve((size_t)N_GRAPHS * 2 * DIM * 4);
    float*    z1   = (float*)carve((size_t)N_GRAPHS * DIM * 4);
    float*    za   = (float*)carve((size_t)N_GRAPHS * DIM * 4);
    float*    zb   = (float*)carve((size_t)N_GRAPHS * (DIM / 2) * 4);
    float*    zc   = (float*)carve((size_t)N_GRAPHS * (DIM / 4) * 4);
    _Float16* w1p  = (_Float16*)carve((size_t)NLAYERS * DIM * 2 * DIM * 2);
    _Float16* w2p  = (_Float16*)carve((size_t)NLAYERS * 2 * DIM * DIM * 2);
    _Float16* sW2p = (_Float16*)carve((size_t)DIM * DIM * 2);
    _Float16* oW0p = (_Float16*)carve((size_t)2 * DIM * DIM * 2);
    _Float16* oW1p = (_Float16*)carve((size_t)DIM * (DIM / 2) * 2);
    _Float16* oW2p = (_Float16*)carve((size_t)(DIM / 2) * (DIM / 4) * 2);

    // ---- weight packing into WMMA fragment order (f16) -------------------
    for (int l = 0; l < NLAYERS; ++l) {
        int n1 = DIM * 2 * DIM;
        pack_w_f16_kernel<<<(n1 + 255) / 256, 256, 0, stream>>>(
            mlp1_W + (size_t)l * n1, w1p + (size_t)l * n1, DIM, 2 * DIM);
        pack_w_f16_kernel<<<(n1 + 255) / 256, 256, 0, stream>>>(
            mlp2_W + (size_t)l * n1, w2p + (size_t)l * n1, 2 * DIM, DIM);
    }
    pack_w_f16_kernel<<<(DIM * DIM + 255) / 256, 256, 0, stream>>>(
        sW2, sW2p, DIM, DIM);
    pack_w_f16_kernel<<<(2 * DIM * DIM + 255) / 256, 256, 0, stream>>>(
        out_W0, oW0p, 2 * DIM, DIM);
    pack_w_f16_kernel<<<(DIM * (DIM / 2) + 255) / 256, 256, 0, stream>>>(
        out_W1, oW1p, DIM, DIM / 2);
    pack_w_f16_kernel<<<((DIM / 2) * (DIM / 4) + 255) / 256, 256, 0, stream>>>(
        out_W2, oW2p, DIM / 2, DIM / 4);

    // ---- atom embedding: h = relu(solute_x @ x_emb_W + b) ----------------
    {
        int total = N_NODES * DIM;
        dense_smallk_relu_kernel<<<(total + 255) / 256, 256, 0, stream>>>(
            solute_x, x_emb_W, x_emb_b, h, N_NODES, DIM, F_ATOM);
    }

    // ---- GNN layers ------------------------------------------------------
    const int total_hD = N_NODES * DIM;
    for (int l = 0; l < NLAYERS; ++l) {
        const float* e1l = edge_emb1 + (size_t)l * 6 * DIM;
        const float* e2l = edge_emb2 + (size_t)l * 3 * DIM;

        agg_init_kernel<<<(total_hD + 255) / 256, 256, 0, stream>>>(
            h, e1l, e2l, agg, total_hD);
        edge_scatter_kernel<<<N_EDGES, 256, 0, stream>>>(
            h, e_src, e_dst, edge_attr, e1l, e2l, agg);

        // mid = relu(agg @ W1 + b1) : M=60000, N=1024, K=512 (TN=4 -> 256 cols/blk)
        gemm_wmma_tdm_kernel<4><<<dim3(N_NODES / 16, (2 * DIM) / 256), 128, 0, stream>>>(
            agg, DIM, w1p + (size_t)l * DIM * 2 * DIM, mlp1_b + (size_t)l * 2 * DIM,
            mid, 2 * DIM, 2 * DIM, DIM, 1);

        // h2(=agg) = mid @ W2 + b2 : M=60000, N=512, K=1024
        gemm_wmma_tdm_kernel<4><<<dim3(N_NODES / 16, DIM / 256), 128, 0, stream>>>(
            mid, 2 * DIM, w2p + (size_t)l * 2 * DIM * DIM, mlp2_b + (size_t)l * DIM,
            agg, DIM, DIM, 2 * DIM, 0);

        // batchnorm (+relu except last layer)
        bn_stats_kernel<<<DIM, 256, 0, stream>>>(agg, stats, N_NODES, DIM);
        bn_apply_kernel<<<(total_hD + 255) / 256, 256, 0, stream>>>(
            agg, stats, bn_gamma + (size_t)l * DIM, bn_beta + (size_t)l * DIM,
            h, total_hD, (l < NLAYERS - 1) ? 1 : 0);
    }

    // ---- readout ---------------------------------------------------------
    {
        int zt = N_GRAPHS * 2 * DIM;
        zero_kernel<<<(zt + 255) / 256, 256, 0, stream>>>(zcat, zt);
    }
    pool_kernel<<<N_NODES, 256, 0, stream>>>(h, batch, zcat);

    // solvent branch: z1 = relu(solvent_x @ sW1 + sb1); zcat[:,512:] = z1 @ sW2 + sb2
    {
        int total = N_GRAPHS * DIM;
        dense_smallk_relu_kernel<<<(total + 255) / 256, 256, 0, stream>>>(
            solvent_x, sW1, sb1, z1, N_GRAPHS, DIM, F_ATOM);
        gemm_wmma_tdm_kernel<4><<<dim3(N_GRAPHS / 16, DIM / 256), 128, 0, stream>>>(
            z1, DIM, sW2p, sb2, zcat + DIM, 2 * DIM, DIM, DIM, 0);
    }

    // output MLP stack
    gemm_wmma_tdm_kernel<4><<<dim3(N_GRAPHS / 16, DIM / 256), 128, 0, stream>>>(
        zcat, 2 * DIM, oW0p, out_b0, za, DIM, DIM, 2 * DIM, 1);
    gemm_wmma_tdm_kernel<4><<<dim3(N_GRAPHS / 16, (DIM / 2) / 256), 128, 0, stream>>>(
        za, DIM, oW1p, out_b1, zb, DIM / 2, DIM / 2, DIM, 1);
    gemm_wmma_tdm_kernel<1><<<dim3(N_GRAPHS / 16, (DIM / 4) / 64), 128, 0, stream>>>(
        zb, DIM / 2, oW2p, out_b2, zc, DIM / 4, DIM / 4, DIM / 2, 1);

    final_dot_kernel<<<(N_GRAPHS + 255) / 256, 256, 0, stream>>>(
        zc, last_W, last_b, (float*)d_out, N_GRAPHS, DIM / 4);
}
